// TransformerBlock_76819785056635
// MI455X (gfx1250) — matrix operand order repaired and checked by reading, not compiled, not run on hardware
//
#include <hip/hip_runtime.h>
#include <math.h>

// ---------------------------------------------------------------------------
// Problem constants (from the reference)
// ---------------------------------------------------------------------------
#define DIMC   128
#define HEADS  8
#define SEQ    512
#define NEXP   8
#define HIDC   512
#define T_TOK  4096            // B*HEADS*SEQ
#define TD     (T_TOK * DIMC)  // 524288 floats
#define SCALE_ATTN 0.08838834764831845f  // 128^-0.5

typedef __attribute__((ext_vector_type(16))) _Float16 v16h;
typedef __attribute__((ext_vector_type(8)))  float    v8f;

// ---------------------------------------------------------------------------
// Complex NT GEMM:  C[M,N] = A[M,K] (complex) * W[N,K]^T (complex)
//   out_r = Ar.Br + s2 * Ai.Bi
//   out_i = Ai.Br + s3 * Ar.Bi
//   clinear: s2=-1, s3=+1    |    attention scores: s2=+1, s3=-1
// mode 0: write Cr,Ci (+bias, +optional elementwise add, e.g. PE)
// mode 1: write Cr = sqrt(cr^2+ci^2+1e-8)*scale   (fused score magnitude)
// mode 2: Cr += rowScale[m]*(cr+biasR), Ci += rowScale[m]*(ci+biasI)  (MoE fc2)
// Double-buffered LDS: global loads of tile k+1 issue before the WMMAs of
// tile k; one barrier per K-step.
// ---------------------------------------------------------------------------
#define BM 64
#define BN 64
#define BK 32
#define LDT 40  // padded half stride

__global__ __launch_bounds__(256) void cgemm_nt(
    const float* __restrict__ Ar, const float* __restrict__ Ai, long sA,
    const float* __restrict__ Br, const float* __restrict__ Bi, long sB,
    const float* __restrict__ biasR, const float* __restrict__ biasI,
    const float* __restrict__ addR, const float* __restrict__ addI,
    float* __restrict__ Cr, float* __restrict__ Ci, long sC,
    const float* __restrict__ rowScale, int rowScaleStride,
    int M, int N, int K, float s2, float s3, int mode, float scale)
{
  __shared__ _Float16 As[2][2][BM][LDT];  // [buf][r/i][m][k]
  __shared__ _Float16 Bs[2][2][BN][LDT];

  const int tid  = threadIdx.x;
  const int lane = tid & 31;
  const int wave = tid >> 5;
  const int wm   = wave & 1;   // 2 wave rows (M)
  const int wn   = wave >> 1;  // 4 wave cols (N)
  const int bm   = blockIdx.x * BM;
  const int bn   = blockIdx.y * BN;
  const long bz  = blockIdx.z;

  Ar += bz * sA; Ai += bz * sA;
  Br += bz * sB; Bi += bz * sB;
  Cr += bz * sC; if (Ci)   Ci   += bz * sC;
  if (addR) { addR += bz * sC; addI += bz * sC; }

  v8f accRR[2], accII[2], accIR[2], accRI[2];
#pragma unroll
  for (int s = 0; s < 2; ++s) {
    accRR[s] = (v8f){0,0,0,0,0,0,0,0};
    accII[s] = (v8f){0,0,0,0,0,0,0,0};
    accIR[s] = (v8f){0,0,0,0,0,0,0,0};
    accRI[s] = (v8f){0,0,0,0,0,0,0,0};
  }

  const int lr = tid >> 2;        // 64 rows per tile
  const int lc = (tid & 3) * 8;   // 8 consecutive k per thread
  const int kh = lane >> 4;       // 0/1: K half within fragments
  const int ml = lane & 15;

  // Branchless ragged-N handling (router N=8): clamp row, mask values.
  const int   nrow  = bn + lr;
  const int   nrowc = (nrow < N) ? nrow : (N - 1);
  const float bmask = (nrow < N) ? 1.f : 0.f;

  const float* Arow_r = Ar + (long)(bm + lr) * K + lc;
  const float* Arow_i = Ai + (long)(bm + lr) * K + lc;
  const float* Brow_r = Br + (long)nrowc * K + lc;
  const float* Brow_i = Bi + (long)nrowc * K + lc;

  float ra[8], rai[8], rb[8], rbi[8];

  auto gload = [&](int k0) {
#pragma unroll
    for (int j = 0; j < 8; ++j) {
      ra[j]  = Arow_r[k0 + j];
      rai[j] = Arow_i[k0 + j];
      rb[j]  = Brow_r[k0 + j] * bmask;
      rbi[j] = Brow_i[k0 + j] * bmask;
    }
  };
  auto sstore = [&](int buf) {
#pragma unroll
    for (int j = 0; j < 8; ++j) {
      As[buf][0][lr][lc + j] = (_Float16)ra[j];
      As[buf][1][lr][lc + j] = (_Float16)rai[j];
      Bs[buf][0][lr][lc + j] = (_Float16)rb[j];
      Bs[buf][1][lr][lc + j] = (_Float16)rbi[j];
    }
  };

  gload(0);
  sstore(0);
  int cur = 0;

  for (int k0 = 0; k0 < K; k0 += BK) {
    __syncthreads();  // buffer `cur` staged

    const bool more = (k0 + BK) < K;
    if (more) gload(k0 + BK);          // overlaps with WMMAs below
    if (k0 + 2 * BK < K) {             // gfx1250 global_prefetch_b8
      __builtin_prefetch(Arow_r + k0 + 2 * BK, 0, 1);
      __builtin_prefetch(Arow_i + k0 + 2 * BK, 0, 1);
      __builtin_prefetch(Brow_r + k0 + 2 * BK, 0, 1);
      __builtin_prefetch(Brow_i + k0 + 2 * BK, 0, 1);
    }

    const _Float16 (*A0)[LDT] = As[cur][0];
    const _Float16 (*A1)[LDT] = As[cur][1];
    const _Float16 (*B0)[LDT] = Bs[cur][0];
    const _Float16 (*B1)[LDT] = Bs[cur][1];

    // ---- B fragments (16-bit B 32x16, same k order as A: lanes0-15 K0..7,16..23; lanes16-31 K8..15,24..31) ----
    v16h brf, bif;
    {
      _Float16* b0 = (_Float16*)&brf;
      _Float16* b1 = (_Float16*)&bif;
      const int nn = wn * 16 + ml;
#pragma unroll
      for (int v = 0; v < 8; ++v) {
        const int kk = (v < 4) ? (2 * v + 8 * kh) : (16 + 2 * (v - 4) + 8 * kh);
        b0[2*v]   = B0[nn][kk];
        b0[2*v+1] = B0[nn][kk+1];
        b1[2*v]   = B1[nn][kk];
        b1[2*v+1] = B1[nn][kk+1];
      }
    }
    // ---- A fragments (ISA 16-bit A 16x32) + 4 WMMAs per subtile ----
#pragma unroll
    for (int s = 0; s < 2; ++s) {
      v16h arf, aif;
      _Float16* a0 = (_Float16*)&arf;
      _Float16* a1 = (_Float16*)&aif;
      const int mm = wm * 32 + s * 16 + ml;
#pragma unroll
      for (int v = 0; v < 8; ++v) {
        const int kk = (v < 4) ? (2 * v + 8 * kh) : (16 + 2 * (v - 4) + 8 * kh);
        a0[2*v]   = A0[mm][kk];
        a0[2*v+1] = A0[mm][kk+1];
        a1[2*v]   = A1[mm][kk];
        a1[2*v+1] = A1[mm][kk+1];
      }
      accRR[s] = __builtin_amdgcn_wmma_f32_16x16x32_f16(false, arf, false, brf, (short)0, accRR[s], false, false);
      accII[s] = __builtin_amdgcn_wmma_f32_16x16x32_f16(false, aif, false, bif, (short)0, accII[s], false, false);
      accIR[s] = __builtin_amdgcn_wmma_f32_16x16x32_f16(false, aif, false, brf, (short)0, accIR[s], false, false);
      accRI[s] = __builtin_amdgcn_wmma_f32_16x16x32_f16(false, arf, false, bif, (short)0, accRI[s], false, false);
    }

    if (more) sstore(cur ^ 1);  // stage next buffer; barrier at loop top protects
    cur ^= 1;
  }

  // ---- epilogue (C layout: VGPR r -> M = base + r + 8*(lane/16), N = lane%16) ----
  const int nn = bn + wn * 16 + ml;
  if (nn < N) {
    const float bR = biasR ? biasR[nn] : 0.f;
    const float bI = biasI ? biasI[nn] : 0.f;
#pragma unroll
    for (int s = 0; s < 2; ++s) {
#pragma unroll
      for (int r = 0; r < 8; ++r) {
        const int mm = bm + wm * 32 + s * 16 + kh * 8 + r;
        float cr = accRR[s][r] + s2 * accII[s][r] + bR;
        float ci = accIR[s][r] + s3 * accRI[s][r] + bI;
        const long idx = (long)mm * N + nn;
        if (mode == 0) {
          if (addR) { cr += addR[idx]; ci += addI[idx]; }
          Cr[idx] = cr; Ci[idx] = ci;
        } else if (mode == 1) {
          Cr[idx] = sqrtf(cr * cr + ci * ci + 1e-8f) * scale;
        } else {
          const float w = rowScale[(long)mm * rowScaleStride];
          Cr[idx] += w * cr;
          Ci[idx] += w * ci;
        }
      }
    }
  }
}

// ---------------------------------------------------------------------------
// Real x Complex NN GEMM:  C[M,N] = A[M,K] (real, softmax probs) * V[K,N] (complex)
// Double-buffered like cgemm_nt. B staged transposed into LDS: Bs[n][k].
// ---------------------------------------------------------------------------
__global__ __launch_bounds__(256) void rgemm_nn(
    const float* __restrict__ A, long sA,
    const float* __restrict__ Br, const float* __restrict__ Bi, long sB,
    float* __restrict__ Cr, float* __restrict__ Ci, long sC,
    int M, int N, int K)
{
  __shared__ _Float16 As[2][BM][LDT];
  __shared__ _Float16 Bs[2][2][BN][LDT];  // [buf][r/i][n][k]

  const int tid  = threadIdx.x;
  const int lane = tid & 31;
  const int wave = tid >> 5;
  const int wm   = wave & 1;
  const int wn   = wave >> 1;
  const int bm   = blockIdx.x * BM;
  const int bn   = blockIdx.y * BN;
  const long bz  = blockIdx.z;

  A  += bz * sA;
  Br += bz * sB; Bi += bz * sB;
  Cr += bz * sC; Ci += bz * sC;

  v8f accR[2], accI[2];
#pragma unroll
  for (int s = 0; s < 2; ++s) {
    accR[s] = (v8f){0,0,0,0,0,0,0,0};
    accI[s] = (v8f){0,0,0,0,0,0,0,0};
  }

  const int lr = tid >> 2;
  const int lc = (tid & 3) * 8;
  const int bk_row = tid >> 3;       // 32 k rows
  const int bn0    = (tid & 7) * 8;  // 8 consecutive n
  const int kh = lane >> 4;
  const int ml = lane & 15;

  const float* Arow  = A  + (long)(bm + lr) * K + lc;
  const float* Bptr_r = Br + (long)bk_row * N + bn + bn0;
  const float* Bptr_i = Bi + (long)bk_row * N + bn + bn0;

  float ra[8], rb[8], rbi[8];

  auto gload = [&](int k0) {
#pragma unroll
    for (int j = 0; j < 8; ++j) {
      ra[j]  = Arow[k0 + j];
      rb[j]  = Bptr_r[(long)k0 * N + j];
      rbi[j] = Bptr_i[(long)k0 * N + j];
    }
  };
  auto sstore = [&](int buf) {
#pragma unroll
    for (int j = 0; j < 8; ++j) {
      As[buf][lr][lc + j] = (_Float16)ra[j];
      Bs[buf][0][bn0 + j][bk_row] = (_Float16)rb[j];
      Bs[buf][1][bn0 + j][bk_row] = (_Float16)rbi[j];
    }
  };

  gload(0);
  sstore(0);
  int cur = 0;

  for (int k0 = 0; k0 < K; k0 += BK) {
    __syncthreads();

    const bool more = (k0 + BK) < K;
    if (more) gload(k0 + BK);
    if (k0 + 2 * BK < K) {
      __builtin_prefetch(Arow + k0 + 2 * BK, 0, 1);
      __builtin_prefetch(Bptr_r + (long)(k0 + 2 * BK) * N, 0, 1);
      __builtin_prefetch(Bptr_i + (long)(k0 + 2 * BK) * N, 0, 1);
    }

    const _Float16 (*A0)[LDT] = As[cur];
    const _Float16 (*B0)[LDT] = Bs[cur][0];
    const _Float16 (*B1)[LDT] = Bs[cur][1];

    v16h brf, bif;
    {
      _Float16* b0 = (_Float16*)&brf;
      _Float16* b1 = (_Float16*)&bif;
      const int nn = wn * 16 + ml;
#pragma unroll
      for (int v = 0; v < 8; ++v) {
        const int kk = (v < 4) ? (2 * v + 8 * kh) : (16 + 2 * (v - 4) + 8 * kh);
        b0[2*v]   = B0[nn][kk];
        b0[2*v+1] = B0[nn][kk+1];
        b1[2*v]   = B1[nn][kk];
        b1[2*v+1] = B1[nn][kk+1];
      }
    }
#pragma unroll
    for (int s = 0; s < 2; ++s) {
      v16h af;
      _Float16* a0 = (_Float16*)&af;
      const int mm = wm * 32 + s * 16 + ml;
#pragma unroll
      for (int v = 0; v < 8; ++v) {
        const int kk = (v < 4) ? (2 * v + 8 * kh) : (16 + 2 * (v - 4) + 8 * kh);
        a0[2*v]   = A0[mm][kk];
        a0[2*v+1] = A0[mm][kk+1];
      }
      accR[s] = __builtin_amdgcn_wmma_f32_16x16x32_f16(false, af, false, brf, (short)0, accR[s], false, false);
      accI[s] = __builtin_amdgcn_wmma_f32_16x16x32_f16(false, af, false, bif, (short)0, accI[s], false, false);
    }

    if (more) sstore(cur ^ 1);
    cur ^= 1;
  }

  const int nn = bn + wn * 16 + ml;
  if (nn < N) {
#pragma unroll
    for (int s = 0; s < 2; ++s) {
#pragma unroll
      for (int r = 0; r < 8; ++r) {
        const int mm = bm + wm * 32 + s * 16 + kh * 8 + r;
        const long idx = (long)mm * N + nn;
        Cr[idx] = accR[s][r];
        Ci[idx] = accI[s][r];
      }
    }
  }
}

// ---------------------------------------------------------------------------
// Row softmax, in place. One block per row.
// ---------------------------------------------------------------------------
__global__ __launch_bounds__(256) void softmax_rows(float* __restrict__ x, int L)
{
  __shared__ float red[256];
  float* row = x + (long)blockIdx.x * L;
  const int tid = threadIdx.x;

  float m = -1e30f;
  for (int i = tid; i < L; i += 256) m = fmaxf(m, row[i]);
  red[tid] = m; __syncthreads();
  for (int s = 128; s > 0; s >>= 1) { if (tid < s) red[tid] = fmaxf(red[tid], red[tid + s]); __syncthreads(); }
  m = red[0]; __syncthreads();

  float sum = 0.f;
  for (int i = tid; i < L; i += 256) { const float e = __expf(row[i] - m); row[i] = e; sum += e; }
  red[tid] = sum; __syncthreads();
  for (int s = 128; s > 0; s >>= 1) { if (tid < s) red[tid] += red[tid + s]; __syncthreads(); }
  const float inv = 1.f / red[0];
  for (int i = tid; i < L; i += 256) row[i] *= inv;
}

// ---------------------------------------------------------------------------
// Complex LayerNorm (magnitude LN, phase preserved). One block (128 thr) / token.
// x = (xr+ar, xi+ai); residual pair may be null.
// ---------------------------------------------------------------------------
__global__ __launch_bounds__(128) void cln_k(
    const float* __restrict__ xr, const float* __restrict__ xi,
    const float* __restrict__ ar, const float* __restrict__ ai,
    const float* __restrict__ gamma, const float* __restrict__ betaR,
    const float* __restrict__ betaI,
    float* __restrict__ outR, float* __restrict__ outI)
{
  __shared__ float red[128];
  const int d = threadIdx.x;
  const long idx = (long)blockIdx.x * DIMC + d;

  float r  = xr[idx] + (ar ? ar[idx] : 0.f);
  float im = xi[idx] + (ai ? ai[idx] : 0.f);
  const float mag = sqrtf(r * r + im * im + 1e-5f);

  red[d] = mag; __syncthreads();
  for (int s = 64; s > 0; s >>= 1) { if (d < s) red[d] += red[d + s]; __syncthreads(); }
  const float mean = red[0] * (1.f / DIMC); __syncthreads();

  const float dev = mag - mean;
  red[d] = dev * dev; __syncthreads();
  for (int s = 64; s > 0; s >>= 1) { if (d < s) red[d] += red[d + s]; __syncthreads(); }
  const float stdv = sqrtf(red[0] * (1.f / (DIMC - 1))) + 1e-5f;

  const float nm = dev / stdv * gamma[d];
  const float h  = sqrtf(r * r + im * im);
  const float cs = (h > 0.f) ? (r / h)  : 1.f;   // cos(atan2(0,0)) = 1
  const float sn = (h > 0.f) ? (im / h) : 0.f;
  outR[idx] = nm * cs + betaR[d];
  outI[idx] = nm * sn + betaI[d];
}

// ---------------------------------------------------------------------------
// Elementwise: complex gate multiply  gd = g * ao
// ---------------------------------------------------------------------------
__global__ void cmul_k(const float* __restrict__ gr, const float* __restrict__ gi,
                       const float* __restrict__ aor, const float* __restrict__ aoi,
                       float* __restrict__ outR, float* __restrict__ outI, long n)
{
  const long i = (long)blockIdx.x * blockDim.x + threadIdx.x;
  if (i >= n) return;
  const float a = gr[i], b = gi[i], c = aor[i], d = aoi[i];
  outR[i] = a * c - b * d;
  outI[i] = a * d + b * c;
}

// ---------------------------------------------------------------------------
// Elementwise cSwiGLU: g <- polar(sigmoid(|g|), arg(g)) * u   (in place on g)
// ---------------------------------------------------------------------------
__global__ void cswiglu_k(float* __restrict__ gr, float* __restrict__ gi,
                          const float* __restrict__ ur, const float* __restrict__ ui, long n)
{
  const long i = (long)blockIdx.x * blockDim.x + threadIdx.x;
  if (i >= n) return;
  const float a = gr[i], b = gi[i];
  const float m2 = a * a + b * b;
  const float sg = 1.f / (1.f + __expf(-sqrtf(m2 + 1e-8f)));
  const float h  = sqrtf(m2);
  const float cs = (h > 0.f) ? (a / h) : 1.f;
  const float sn = (h > 0.f) ? (b / h) : 0.f;
  const float g2r = sg * cs, g2i = sg * sn;
  const float c = ur[i], d = ui[i];
  gr[i] = g2r * c - g2i * d;
  gi[i] = g2r * d + g2i * c;
}

// ---------------------------------------------------------------------------
// Router: |logits| -> top-2 -> softmax -> dense comb weights [T,8]
// ---------------------------------------------------------------------------
__global__ void router_topk(const float* __restrict__ lr, const float* __restrict__ li,
                            float* __restrict__ comb, int T)
{
  const int t = blockIdx.x * blockDim.x + threadIdx.x;
  if (t >= T) return;
  float mag[NEXP];
#pragma unroll
  for (int e = 0; e < NEXP; ++e) {
    const float a = lr[t * NEXP + e], b = li[t * NEXP + e];
    mag[e] = sqrtf(a * a + b * b);
  }
  int i0 = 0;
#pragma unroll
  for (int e = 1; e < NEXP; ++e) if (mag[e] > mag[i0]) i0 = e;
  int i1 = (i0 == 0) ? 1 : 0;
#pragma unroll
  for (int e = 0; e < NEXP; ++e) if (e != i0 && mag[e] > mag[i1]) i1 = e;
  const float e1 = __expf(mag[i1] - mag[i0]);
  const float z  = 1.f + e1;
#pragma unroll
  for (int e = 0; e < NEXP; ++e) comb[t * NEXP + e] = 0.f;
  comb[t * NEXP + i0] = 1.f / z;
  comb[t * NEXP + i1] = e1 / z;
}

__global__ void zero_k(float* __restrict__ p, long n)
{
  const long i = (long)blockIdx.x * blockDim.x + threadIdx.x;
  if (i < n) p[i] = 0.f;
}

// ---------------------------------------------------------------------------
// Host-side orchestration
// ---------------------------------------------------------------------------
struct AttnP {
  const float *q_wr, *q_wi, *q_br, *q_bi;
  const float *k_wr, *k_wi, *k_br, *k_bi;
  const float *v_wr, *v_wi, *v_br, *v_bi;
  const float *g_wr, *g_wi, *g_br, *g_bi;
  const float *o_wr, *o_wi, *o_br, *o_bi;
};
struct MoeP {
  const float *rt_wr, *rt_wi;
  const float *fc1_wr, *fc1_wi, *fc1_br, *fc1_bi;   // [E,HID,DIM] / [E,HID]
  const float *fc2_wr, *fc2_wi, *fc2_br, *fc2_bi;   // [E,DIM,HID] / [E,DIM]
  const float *ag_wr, *ag_wi;                       // [E,HID,HID]
  const float *au_wr, *au_wi;                       // [E,HID,HID]
};
struct LnP { const float *gamma, *betaR, *betaI; };

static void launch_cgemm(hipStream_t st,
    const float* Ar, const float* Ai, long sA,
    const float* Br, const float* Bi, long sB,
    const float* bR, const float* bI,
    const float* addR, const float* addI,
    float* Cr, float* Ci, long sC,
    const float* rowScale, int rsStride,
    int M, int N, int K, int batch, float s2, float s3, int mode, float scale)
{
  dim3 g((M + BM - 1) / BM, (N + BN - 1) / BN, batch);
  cgemm_nt<<<g, 256, 0, st>>>(Ar, Ai, sA, Br, Bi, sB, bR, bI, addR, addI,
                              Cr, Ci, sC, rowScale, rsStride, M, N, K, s2, s3, mode, scale);
}

static void run_attention(hipStream_t st, const AttnP& p,
    const float* qxr, const float* qxi,
    const float* kxr, const float* kxi,
    const float* peqr, const float* peqi,
    const float* pekr, const float* peki,
    float* outR, float* outI, float* arena)
{
  float* qp_r = arena + 0L * TD;  float* qp_i = arena + 1L * TD;
  float* kp_r = arena + 2L * TD;  float* kp_i = arena + 3L * TD;
  float* vp_r = arena + 4L * TD;  float* vp_i = arena + 5L * TD;
  float* gp_r = arena + 6L * TD;  float* gp_i = arena + 7L * TD;
  float* ao_r = arena + 8L * TD;  float* ao_i = arena + 9L * TD;
  float* gd_r = arena + 10L * TD; float* gd_i = arena + 11L * TD;
  float* scor = arena + 12L * TD;                 // [H, S, S] = 4*TD floats

  // projections (clinear signs: s2=-1, s3=+1), PE fused into q/k epilogue
  launch_cgemm(st, qxr, qxi, 0, p.q_wr, p.q_wi, 0, p.q_br, p.q_bi, peqr, peqi,
               qp_r, qp_i, 0, nullptr, 0, T_TOK, DIMC, DIMC, 1, -1.f, 1.f, 0, 1.f);
  launch_cgemm(st, kxr, kxi, 0, p.k_wr, p.k_wi, 0, p.k_br, p.k_bi, pekr, peki,
               kp_r, kp_i, 0, nullptr, 0, T_TOK, DIMC, DIMC, 1, -1.f, 1.f, 0, 1.f);
  launch_cgemm(st, kxr, kxi, 0, p.v_wr, p.v_wi, 0, p.v_br, p.v_bi, nullptr, nullptr,
               vp_r, vp_i, 0, nullptr, 0, T_TOK, DIMC, DIMC, 1, -1.f, 1.f, 0, 1.f);
  launch_cgemm(st, qxr, qxi, 0, p.g_wr, p.g_wi, 0, p.g_br, p.g_bi, nullptr, nullptr,
               gp_r, gp_i, 0, nullptr, 0, T_TOK, DIMC, DIMC, 1, -1.f, 1.f, 0, 1.f);

  // scores per head: sr = Qr.Kr + Qi.Ki ; si = Qi.Kr - Qr.Ki ; fused |.|*SCALE
  launch_cgemm(st, qp_r, qp_i, (long)SEQ * DIMC, kp_r, kp_i, (long)SEQ * DIMC,
               nullptr, nullptr, nullptr, nullptr,
               scor, nullptr, (long)SEQ * SEQ, nullptr, 0,
               SEQ, SEQ, DIMC, HEADS, 1.f, -1.f, 1, SCALE_ATTN);

  softmax_rows<<<HEADS * SEQ, 256, 0, st>>>(scor, SEQ);

  // AV per head: probs[S,S] x V[S,D] (complex)
  {
    dim3 g(SEQ / BM, DIMC / BN, HEADS);
    rgemm_nn<<<g, 256, 0, st>>>(scor, (long)SEQ * SEQ, vp_r, vp_i, (long)SEQ * DIMC,
                                ao_r, ao_i, (long)SEQ * DIMC, SEQ, DIMC, SEQ);
  }

  cmul_k<<<(TD + 255) / 256, 256, 0, st>>>(gp_r, gp_i, ao_r, ao_i, gd_r, gd_i, TD);

  launch_cgemm(st, gd_r, gd_i, 0, p.o_wr, p.o_wi, 0, p.o_br, p.o_bi, nullptr, nullptr,
               outR, outI, 0, nullptr, 0, T_TOK, DIMC, DIMC, 1, -1.f, 1.f, 0, 1.f);
}

static void run_moe(hipStream_t st, const MoeP& p,
    const float* xr, const float* xi,
    float* accR, float* accI, float* arena)
{
  const long THID = (long)T_TOK * HIDC;  // 4*TD
  float* hr  = arena + 0L  * TD;
  float* hi  = arena + 4L  * TD;
  float* g2r = arena + 8L  * TD;
  float* g2i = arena + 12L * TD;
  float* u2r = arena + 16L * TD;
  float* u2i = arena + 20L * TD;
  float* lr_ = arena + 24L * TD;
  float* li_ = lr_ + (long)T_TOK * NEXP;
  float* comb = li_ + (long)T_TOK * NEXP;

  zero_k<<<(TD + 255) / 256, 256, 0, st>>>(accR, TD);
  zero_k<<<(TD + 255) / 256, 256, 0, st>>>(accI, TD);

  // router (no bias), N = 8 (ragged, masked in kernel)
  launch_cgemm(st, xr, xi, 0, p.rt_wr, p.rt_wi, 0, nullptr, nullptr, nullptr, nullptr,
               lr_, li_, 0, nullptr, 0, T_TOK, NEXP, DIMC, 1, -1.f, 1.f, 0, 1.f);
  router_topk<<<(T_TOK + 255) / 256, 256, 0, st>>>(lr_, li_, comb, T_TOK);

  for (int e = 0; e < NEXP; ++e) {
    const long wo1 = (long)e * HIDC * DIMC;
    const long wo2 = (long)e * DIMC * HIDC;
    const long woA = (long)e * HIDC * HIDC;
    // fc1
    launch_cgemm(st, xr, xi, 0, p.fc1_wr + wo1, p.fc1_wi + wo1, 0,
                 p.fc1_br + (long)e * HIDC, p.fc1_bi + (long)e * HIDC, nullptr, nullptr,
                 hr, hi, 0, nullptr, 0, T_TOK, HIDC, DIMC, 1, -1.f, 1.f, 0, 1.f);
    // act_gate / act_up (no bias)
    launch_cgemm(st, hr, hi, 0, p.ag_wr + woA, p.ag_wi + woA, 0, nullptr, nullptr,
                 nullptr, nullptr, g2r, g2i, 0, nullptr, 0,
                 T_TOK, HIDC, HIDC, 1, -1.f, 1.f, 0, 1.f);
    launch_cgemm(st, hr, hi, 0, p.au_wr + woA, p.au_wi + woA, 0, nullptr, nullptr,
                 nullptr, nullptr, u2r, u2i, 0, nullptr, 0,
                 T_TOK, HIDC, HIDC, 1, -1.f, 1.f, 0, 1.f);
    cswiglu_k<<<(int)((THID + 255) / 256), 256, 0, st>>>(g2r, g2i, u2r, u2i, THID);
    // fc2 with fused top-k combine: acc += comb[t,e] * (gemm + bias)
    launch_cgemm(st, g2r, g2i, 0, p.fc2_wr + wo2, p.fc2_wi + wo2, 0,
                 p.fc2_br + (long)e * DIMC, p.fc2_bi + (long)e * DIMC, nullptr, nullptr,
                 accR, accI, 0, comb + e, NEXP,
                 T_TOK, DIMC, HIDC, 1, -1.f, 1.f, 2, 1.f);
  }
}

// ---------------------------------------------------------------------------
// kernel_launch
// ---------------------------------------------------------------------------
extern "C" void kernel_launch(void* const* d_in, const int* in_sizes, int n_in,
                              void* d_out, int out_size, void* d_ws, size_t ws_size,
                              hipStream_t stream)
{
  (void)in_sizes; (void)n_in; (void)out_size; (void)ws_size;
#define IN(i) ((const float*)d_in[(i)])

  const float* Xin_r  = IN(0); const float* Xin_i  = IN(1);
  const float* Xout_r = IN(2); const float* Xout_i = IN(3);
  const float* PEin_r = IN(4); const float* PEin_i = IN(5);
  const float* PEout_r= IN(6); const float* PEout_i= IN(7);

  // params: jax tree_flatten leaf order (dict keys sorted recursively), base 8
  // attn block (20 leaves): gate{bi,br,wi,wr} k{..} out{..} q{..} v{..}
  auto attn = [&](int b) {
    AttnP a;
    a.g_bi = IN(b+0);  a.g_br = IN(b+1);  a.g_wi = IN(b+2);  a.g_wr = IN(b+3);
    a.k_bi = IN(b+4);  a.k_br = IN(b+5);  a.k_wi = IN(b+6);  a.k_wr = IN(b+7);
    a.o_bi = IN(b+8);  a.o_br = IN(b+9);  a.o_wi = IN(b+10); a.o_wr = IN(b+11);
    a.q_bi = IN(b+12); a.q_br = IN(b+13); a.q_wi = IN(b+14); a.q_wr = IN(b+15);
    a.v_bi = IN(b+16); a.v_br = IN(b+17); a.v_wi = IN(b+18); a.v_wr = IN(b+19);
    return a;
  };
  // moe block (14 leaves): experts{act_gate{wi,wr},act_up{wi,wr},fc1{bi,br,wi,wr},fc2{bi,br,wi,wr}}, router{wi,wr}
  auto moe = [&](int b) {
    MoeP m;
    m.ag_wi  = IN(b+0);  m.ag_wr  = IN(b+1);
    m.au_wi  = IN(b+2);  m.au_wr  = IN(b+3);
    m.fc1_bi = IN(b+4);  m.fc1_br = IN(b+5);  m.fc1_wi = IN(b+6);  m.fc1_wr = IN(b+7);
    m.fc2_bi = IN(b+8);  m.fc2_br = IN(b+9);  m.fc2_wi = IN(b+10); m.fc2_wr = IN(b+11);
    m.rt_wi  = IN(b+12); m.rt_wr  = IN(b+13);
    return m;
  };
  auto ln = [&](int b) { LnP l; l.betaI = IN(b+0); l.betaR = IN(b+1); l.gamma = IN(b+2); return l; };

  const AttnP crossA   = attn(8);
  const MoeP  ffnIn    = moe(28);
  const MoeP  ffnOut   = moe(42);
  const LnP   norm1In  = ln(56);
  const LnP   norm1Out = ln(59);
  const LnP   norm2In  = ln(62);
  const LnP   norm2Out = ln(65);
  const LnP   norm3In  = ln(68);
  const AttnP selfA    = attn(71);
  const AttnP selfAOut = attn(91);

  float* ws   = (float*)d_ws;
  float* xs_r = ws + 0L * TD;
  float* xs_i = ws + 1L * TD;
  float* a_r  = ws + 2L * TD;
  float* a_i  = ws + 3L * TD;
  float* arena = ws + 4L * TD;    // 24*TD + router buffers

  float* out  = (float*)d_out;    // [4, B,H,S,D]
  float* outInR  = out + 0L * TD;
  float* outInI  = out + 1L * TD;
  float* outOutR = out + 2L * TD;
  float* outOutI = out + 3L * TD;

  // ---- 'in' stream ----
  run_attention(stream, selfA, Xin_r, Xin_i, Xin_r, Xin_i,
                PEin_r, PEin_i, PEin_r, PEin_i, a_r, a_i, arena);
  cln_k<<<T_TOK, 128, 0, stream>>>(Xin_r, Xin_i, a_r, a_i,
                                   norm1In.gamma, norm1In.betaR, norm1In.betaI, xs_r, xs_i);

  run_attention(stream, crossA, xs_r, xs_i, Xout_r, Xout_i,
                PEin_r, PEin_i, PEout_r, PEout_i, a_r, a_i, arena);
  cln_k<<<T_TOK, 128, 0, stream>>>(xs_r, xs_i, a_r, a_i,
                                   norm2In.gamma, norm2In.betaR, norm2In.betaI, xs_r, xs_i);

  run_moe(stream, ffnIn, xs_r, xs_i, a_r, a_i, arena);
  cln_k<<<T_TOK, 128, 0, stream>>>(xs_r, xs_i, a_r, a_i,
                                   norm3In.gamma, norm3In.betaR, norm3In.betaI, outInR, outInI);

  // ---- 'out' stream ----
  run_attention(stream, selfAOut, Xout_r, Xout_i, Xout_r, Xout_i,
                PEout_r, PEout_i, PEout_r, PEout_i, a_r, a_i, arena);
  cln_k<<<T_TOK, 128, 0, stream>>>(Xout_r, Xout_i, a_r, a_i,
                                   norm1Out.gamma, norm1Out.betaR, norm1Out.betaI, xs_r, xs_i);

  run_moe(stream, ffnOut, xs_r, xs_i, a_r, a_i, arena);
  cln_k<<<T_TOK, 128, 0, stream>>>(xs_r, xs_i, a_r, a_i,
                                   norm2Out.gamma, norm2Out.betaR, norm2Out.betaI, outOutR, outOutI);
#undef IN
}
